// MultiHeadAttention_52905407152393
// MI455X (gfx1250) — matrix operand order repaired and checked by reading, not compiled, not run on hardware
//
#include <hip/hip_runtime.h>

// ---------------------------------------------------------------------------
// MultiHeadAttention (no softmax) for MI455X / gfx1250, wave32 + WMMA bf16.
//
// Linearity (no softmax!):  (Q K^T) V = Q (K^T V)  -> per (b,head) 64x64 "M".
// All GEMMs on v_wmma_f32_16x16x32_bf16 (fp32 accum). Big GEMMs stage tiles
// into LDS with the Tensor Data Mover (tensor_load_to_lds, TENSORcnt,
// TDM pad feature = bank-conflict padding), double-buffered.
// ---------------------------------------------------------------------------

typedef __bf16 bf16_t;
typedef __attribute__((ext_vector_type(16))) __bf16 v16bf;
typedef __attribute__((ext_vector_type(8)))  __bf16 v8bf;
typedef __attribute__((ext_vector_type(8)))  float  v8f;
typedef __attribute__((ext_vector_type(4)))  unsigned int u32x4;
typedef __attribute__((ext_vector_type(8)))  int i32x8;
typedef __attribute__((ext_vector_type(4)))  int i32x4;

#define DEVINL __device__ __forceinline__

DEVINL v8f vzero() {
  v8f z;
#pragma unroll
  for (int i = 0; i < 8; ++i) z[i] = 0.0f;
  return z;
}

DEVINL v8f wmma_bf16(v16bf a, v16bf b, v8f c) {
  // (neg_a, A, neg_b, B, c_mod, C, reuse_a, reuse_b)
  return __builtin_amdgcn_wmma_f32_16x16x32_bf16(false, a, false, b,
                                                 (short)0, c, false, false);
}

DEVINL unsigned short f32_to_bf16_bits(float f) {
  unsigned int u = __builtin_bit_cast(unsigned int, f);
  u += 0x7fffu + ((u >> 16) & 1u);   // round-to-nearest-even
  return (unsigned short)(u >> 16);
}

union pack8 { unsigned short us[8]; uint4 v; };
union pack4 { unsigned short us[4]; uint2 v; };

DEVINL v16bf cat16(v8bf lo, v8bf hi) {
  v16bf r;
#pragma unroll
  for (int i = 0; i < 8; ++i) { r[i] = lo[i]; r[i + 8] = hi[i]; }
  return r;
}

// A fragment (16x32 bf16, MxK) from row-major (M x K) source, src pre-offset
// to (row0, k0).  lane<16: row=lane, K=0..7 & 16..23; lane>=16: K=8..15 & 24..31.
DEVINL v16bf load_a_frag(const bf16_t* __restrict__ src, int ld, int lane) {
  const int r = lane & 15, half = lane >> 4;
  const bf16_t* p = src + (size_t)r * ld + half * 8;
  return cat16(*(const v8bf*)p, *(const v8bf*)(p + 16));
}

// B fragment (32x16 bf16, KxN) from (N x K) row-major source (B^T rm),
// src pre-offset to (n0, k0).  lane<16: col=lane, K=0..7 & 16..23; lane>=16: K=8..15 & 24..31.
DEVINL v16bf load_b_frag(const bf16_t* __restrict__ src, int ld, int lane) {
  const int nn = lane & 15, half = lane >> 4;
  const bf16_t* p = src + (size_t)nn * ld + half * 8;
  return cat16(*(const v8bf*)p, *(const v8bf*)(p + 16));
}

// Problem constants
static constexpr int Bsz = 4, Tn = 1024, E = 1024, NH = 16, HD = 64;
static constexpr int TOK = Bsz * Tn;       // 4096
static constexpr int E3  = 3 * E;          // 3072
static constexpr int LDS_LD = 40;          // 32 + 8 pad elements per tile row

// ---------------------------------------------------------------------------
// TDM: async-load one 128x32 bf16 tile from a row-major matrix (row stride =
// 1024 elements) into LDS, inserting 16B of padding after every 64B row so
// the LDS image has 40-element rows (bank-conflict-free for ds_load_b128).
// D# bitfields per CDNA5 ISA ch. 8 (group0: count/lds/global/type=2;
// group1: data_size=2B, pad 16dw interval + 4dw amount, dims/strides).
// ---------------------------------------------------------------------------
DEVINL void tdm_load_tile_128x32(unsigned int lds_off, const bf16_t* gsrc) {
  const unsigned long long ga = (unsigned long long)(size_t)gsrc;
  u32x4 g0;
  g0[0] = 1u;                                            // count=1 (valid)
  g0[1] = lds_off;                                       // lds_addr
  g0[2] = (unsigned int)ga;                              // global_addr[31:0]
  g0[3] = ((unsigned int)(ga >> 32) & 0x01FFFFFFu)       // global_addr[56:32]
        | (2u << 30);                                    // type=2 ("image")
  i32x8 g1;
  g1[0] = (1 << 16)       // data_size = 1 -> 2 bytes
        | (1 << 20)       // pad_enable
        | (3 << 22)       // pad_interval: 16 DWORDs (one 64B tile row)
        | (3 << 25);      // pad_amount:   4 DWORDs (16B -> 40-elem pitch)
  g1[1] = (int)(((unsigned)E & 0xFFFFu) << 16);          // tensor_dim0[15:0]
  g1[2] = (int)(((unsigned)E >> 16) | (128u << 16));     // dim0 hi | tensor_dim1=128
  g1[3] = (int)(32u << 16);                              // dim1 hi=0 | tile_dim0=32
  g1[4] = 128;                                           // tile_dim1=128 (tile_dim2=0)
  g1[5] = E;                                             // tensor_dim0_stride lo32
  g1[6] = 0;
  g1[7] = 0;
  i32x4 z4 = (i32x4)0;                                   // groups 2/3 unused (2D)
#if defined(__clang_major__) && (__clang_major__ >= 23)
  i32x8 z8 = (i32x8)0;
  __builtin_amdgcn_tensor_load_to_lds(g0, g1, z4, z4, z8, 0);
#else
  __builtin_amdgcn_tensor_load_to_lds(g0, g1, z4, z4, 0);
#endif
}

// Workspace layout (bytes)
static constexpr size_t OFF_XBF   = 0;                                  // 4096x1024 bf16
static constexpr size_t OFF_WQKV  = OFF_XBF  + (size_t)TOK * E * 2;     // 3072x1024 bf16
static constexpr size_t OFF_WPROJ = OFF_WQKV + (size_t)E3 * E * 2;      // 1024x1024 bf16
static constexpr size_t OFF_QH    = OFF_WPROJ+ (size_t)E * E * 2;       // (b,n,t,h)
static constexpr size_t OFF_KT    = OFF_QH   + (size_t)Bsz*NH*Tn*HD*2;  // (b,n,h,s)
static constexpr size_t OFF_VT    = OFF_KT   + (size_t)Bsz*NH*HD*Tn*2;  // (b,n,j,s)
static constexpr size_t OFF_MT    = OFF_VT   + (size_t)Bsz*NH*HD*Tn*2;  // (b,n,j,i)
static constexpr size_t OFF_ATT   = OFF_MT   + (size_t)Bsz*NH*HD*HD*2;  // (b,t,n*64+j)

// ---------------------------------------------------------------------------
// K0: fp32 -> bf16 conversion (vectorized x4)
// ---------------------------------------------------------------------------
__global__ __launch_bounds__(256) void cvt_bf16_kernel(
    const float* __restrict__ in, bf16_t* __restrict__ out, int n4) {
  int i = blockIdx.x * 256 + threadIdx.x;
  if (i >= n4) return;
  float4 f = ((const float4*)in)[i];
  pack4 p;
  p.us[0] = f32_to_bf16_bits(f.x);
  p.us[1] = f32_to_bf16_bits(f.y);
  p.us[2] = f32_to_bf16_bits(f.z);
  p.us[3] = f32_to_bf16_bits(f.w);
  ((uint2*)out)[i] = p.v;
}

// ---------------------------------------------------------------------------
// K1: QKV = X @ Wqkv^T with fused head-split scatter.
//   128x128 macro-tile; TDM double-buffered LDS staging; 8 waves (2x4),
//   wave tile 64x32.  Q -> Qh (scaled 0.125); K,V -> transposed Kt/Vt.
// ---------------------------------------------------------------------------
__global__ __launch_bounds__(256) void qkv_gemm_kernel(
    const bf16_t* __restrict__ xbf, const bf16_t* __restrict__ wqkv,
    bf16_t* __restrict__ Qh, bf16_t* __restrict__ Kt, bf16_t* __restrict__ Vt) {
  __shared__ __align__(16) bf16_t As[2][128 * LDS_LD];
  __shared__ __align__(16) bf16_t Bs[2][128 * LDS_LD];
  const int tid  = threadIdx.x;
  const int lane = tid & 31, w = tid >> 5;
  const int half = lane >> 4, lr = lane & 15;
  const int m0 = blockIdx.y * 128;   // token rows
  const int n0 = blockIdx.x * 128;   // qkv channels
  const int wm = (w & 1) * 64;
  const int wn = (w >> 1) * 32;

  const bf16_t* gA = xbf  + (size_t)m0 * E;
  const bf16_t* gB = wqkv + (size_t)n0 * E;
  const unsigned int ldsA[2] = {(unsigned int)(size_t)(void*)As[0],
                                (unsigned int)(size_t)(void*)As[1]};
  const unsigned int ldsB[2] = {(unsigned int)(size_t)(void*)Bs[0],
                                (unsigned int)(size_t)(void*)Bs[1]};

  v8f acc[4][2];
#pragma unroll
  for (int mf = 0; mf < 4; ++mf)
#pragma unroll
    for (int nf = 0; nf < 2; ++nf) acc[mf][nf] = vzero();

  if (w == 0) {                      // prologue: stage first tile pair
    tdm_load_tile_128x32(ldsA[0], gA);
    tdm_load_tile_128x32(ldsB[0], gB);
  }
  for (int it = 0; it < E / 32; ++it) {
    if (w == 0) {
      if (it + 1 < E / 32) {
        tdm_load_tile_128x32(ldsA[(it + 1) & 1], gA + (it + 1) * 32);
        tdm_load_tile_128x32(ldsB[(it + 1) & 1], gB + (it + 1) * 32);
        __builtin_amdgcn_s_wait_tensorcnt(2);  // in-order: tile `it` resident
      } else {
        __builtin_amdgcn_s_wait_tensorcnt(0);
      }
    }
    __syncthreads();
    const bf16_t* cA = As[it & 1];
    const bf16_t* cB = Bs[it & 1];
    v16bf a[4], b[2];
#pragma unroll
    for (int mf = 0; mf < 4; ++mf)
      a[mf] = load_a_frag(&cA[(wm + mf * 16) * LDS_LD], LDS_LD, lane);
#pragma unroll
    for (int nf = 0; nf < 2; ++nf)
      b[nf] = load_b_frag(&cB[(wn + nf * 16) * LDS_LD], LDS_LD, lane);
#pragma unroll
    for (int mf = 0; mf < 4; ++mf)
#pragma unroll
      for (int nf = 0; nf < 2; ++nf)
        acc[mf][nf] = wmma_bf16(a[mf], b[nf], acc[mf][nf]);
    __syncthreads();
  }

  // Scatter epilogue. Channel c = sect*1024 + hh*16 + head (head fastest).
  const int bidx = m0 >> 10;  // 128-row tile never straddles a batch
#pragma unroll
  for (int mf = 0; mf < 4; ++mf) {
    const int tbase = (m0 + wm + mf * 16 + half * 8) & 1023;
#pragma unroll
    for (int nf = 0; nf < 2; ++nf) {
      const int c0   = n0 + wn + nf * 16;     // 16-aligned -> head = lr
      const int sect = c0 >> 10;
      const int hh   = (c0 & 1023) >> 4;
      const v8f a8   = acc[mf][nf];
      if (sect == 0) {
        bf16_t* q = Qh + (((size_t)bidx * NH + lr) * Tn + tbase) * HD + hh;
#pragma unroll
        for (int r = 0; r < 8; ++r)
          q[(size_t)r * HD] =
              __builtin_bit_cast(bf16_t, f32_to_bf16_bits(a8[r] * 0.125f));
      } else {
        bf16_t* dst = (sect == 1 ? Kt : Vt) +
                      (((size_t)bidx * NH + lr) * HD + hh) * Tn + tbase;
        pack8 p;
#pragma unroll
        for (int r = 0; r < 8; ++r) p.us[r] = f32_to_bf16_bits(a8[r]);
        *(uint4*)dst = p.v;
      }
    }
  }
}

// ---------------------------------------------------------------------------
// K2: per (b,n):  M(i,j) = sum_s K^T(i,s) V(s,j)  -> store Mt[j][i] (64x64).
// ---------------------------------------------------------------------------
__global__ __launch_bounds__(256) void kv_gemm_kernel(
    const bf16_t* __restrict__ Kt, const bf16_t* __restrict__ Vt,
    bf16_t* __restrict__ Mt) {
  const int bn = blockIdx.x;  // b*16+n
  const int tid = threadIdx.x, lane = tid & 31, w = tid >> 5;
  const int half = lane >> 4, lr = lane & 15;
  const int mf = w >> 1, nf0 = (w & 1) * 2;
  const bf16_t* A  = Kt + (size_t)bn * HD * Tn;  // (i=hh, s) rm
  const bf16_t* Bv = Vt + (size_t)bn * HD * Tn;  // (j,    s) rm  (= B^T rm)
  v8f acc0 = vzero(), acc1 = vzero();
  for (int ks = 0; ks < Tn; ks += 32) {
    if (ks + 256 < Tn) {  // stream-ahead hints -> global_prefetch_b8
      __builtin_prefetch(A  + (size_t)(mf * 16) * Tn + ks + 256, 0, 0);
      __builtin_prefetch(Bv + (size_t)(nf0 * 16) * Tn + ks + 256, 0, 0);
    }
    v16bf a  = load_a_frag(A  + (size_t)(mf * 16) * Tn + ks, Tn, lane);
    v16bf b0 = load_b_frag(Bv + (size_t)(nf0 * 16) * Tn + ks, Tn, lane);
    v16bf b1 = load_b_frag(Bv + (size_t)((nf0 + 1) * 16) * Tn + ks, Tn, lane);
    acc0 = wmma_bf16(a, b0, acc0);
    acc1 = wmma_bf16(a, b1, acc1);
  }
  const int i0 = mf * 16 + half * 8;
#pragma unroll
  for (int q = 0; q < 2; ++q) {
    const int j = (nf0 + q) * 16 + lr;
    const v8f a8 = q ? acc1 : acc0;
    pack8 p;
#pragma unroll
    for (int r = 0; r < 8; ++r) p.us[r] = f32_to_bf16_bits(a8[r]);
    *(uint4*)(Mt + ((size_t)bn * HD + j) * HD + i0) = p.v;  // Mt[j][i0..+7]
  }
}

// ---------------------------------------------------------------------------
// K3: per (b,n): att^T(j,t) = sum_h M^T(j,h) Q(t,h)
//   A = Mt (j x h) rm, B^T = Qh (t x h) rm; transposed C store -> attc.
// ---------------------------------------------------------------------------
__global__ __launch_bounds__(256) void att_kernel(
    const bf16_t* __restrict__ Mt, const bf16_t* __restrict__ Qh,
    bf16_t* __restrict__ attc) {
  const int bn = blockIdx.y;                 // b*16+n
  const int b = bn >> 4, n = bn & 15;
  const int t0 = blockIdx.x * 128;
  const int tid = threadIdx.x, lane = tid & 31, w = tid >> 5;
  const int half = lane >> 4, lr = lane & 15;
  const bf16_t* Am = Mt + (size_t)bn * HD * HD;                    // ld = 64
  const bf16_t* Bq = Qh + ((size_t)bn * Tn + t0 + w * 16) * HD;    // ld = 64
  v8f acc[4];
#pragma unroll
  for (int mf = 0; mf < 4; ++mf) acc[mf] = vzero();
#pragma unroll
  for (int ks = 0; ks < HD; ks += 32) {
    v16bf bfrag = load_b_frag(Bq + ks, HD, lane);
#pragma unroll
    for (int mf = 0; mf < 4; ++mf) {
      v16bf afrag = load_a_frag(Am + (size_t)(mf * 16) * HD + ks, HD, lane);
      acc[mf] = wmma_bf16(afrag, bfrag, acc[mf]);
    }
  }
  const int t = t0 + w * 16 + lr;
  bf16_t* dst = attc + ((size_t)b * Tn + t) * E + n * HD;
#pragma unroll
  for (int mf = 0; mf < 4; ++mf) {
    const int j0 = mf * 16 + half * 8;
    pack8 p;
#pragma unroll
    for (int r = 0; r < 8; ++r) p.us[r] = f32_to_bf16_bits(acc[mf][r]);
    *(uint4*)(dst + j0) = p.v;
  }
}

// ---------------------------------------------------------------------------
// K4: out(t,e') = sum_c attc(t,c) Wproj(e',c), computed as C(e',t) so the
//   fragment-transposed fp32 store hits out[t][e'] contiguously.
//   Same TDM double-buffered staging as K1.
// ---------------------------------------------------------------------------
__global__ __launch_bounds__(256) void proj_gemm_kernel(
    const bf16_t* __restrict__ Wp, const bf16_t* __restrict__ attc,
    float* __restrict__ out) {
  __shared__ __align__(16) bf16_t As[2][128 * LDS_LD];
  __shared__ __align__(16) bf16_t Bs[2][128 * LDS_LD];
  const int b  = blockIdx.z;
  const int e0 = blockIdx.y * 128;   // output feature rows (M)
  const int t0 = blockIdx.x * 128;   // tokens (N)
  const int tid = threadIdx.x, lane = tid & 31, w = tid >> 5;
  const int half = lane >> 4, lr = lane & 15;
  const int wm = (w & 1) * 64, wn = (w >> 1) * 32;

  const bf16_t* gA = Wp + (size_t)e0 * E;
  const bf16_t* gB = attc + (size_t)b * Tn * E + (size_t)t0 * E;
  const unsigned int ldsA[2] = {(unsigned int)(size_t)(void*)As[0],
                                (unsigned int)(size_t)(void*)As[1]};
  const unsigned int ldsB[2] = {(unsigned int)(size_t)(void*)Bs[0],
                                (unsigned int)(size_t)(void*)Bs[1]};

  v8f acc[4][2];
#pragma unroll
  for (int mf = 0; mf < 4; ++mf)
#pragma unroll
    for (int nf = 0; nf < 2; ++nf) acc[mf][nf] = vzero();

  if (w == 0) {
    tdm_load_tile_128x32(ldsA[0], gA);
    tdm_load_tile_128x32(ldsB[0], gB);
  }
  for (int it = 0; it < E / 32; ++it) {
    if (w == 0) {
      if (it + 1 < E / 32) {
        tdm_load_tile_128x32(ldsA[(it + 1) & 1], gA + (it + 1) * 32);
        tdm_load_tile_128x32(ldsB[(it + 1) & 1], gB + (it + 1) * 32);
        __builtin_amdgcn_s_wait_tensorcnt(2);
      } else {
        __builtin_amdgcn_s_wait_tensorcnt(0);
      }
    }
    __syncthreads();
    const bf16_t* cA = As[it & 1];
    const bf16_t* cB = Bs[it & 1];
    v16bf a[4], bb[2];
#pragma unroll
    for (int mf = 0; mf < 4; ++mf)
      a[mf] = load_a_frag(&cA[(wm + mf * 16) * LDS_LD], LDS_LD, lane);
#pragma unroll
    for (int nf = 0; nf < 2; ++nf)
      bb[nf] = load_b_frag(&cB[(wn + nf * 16) * LDS_LD], LDS_LD, lane);
#pragma unroll
    for (int mf = 0; mf < 4; ++mf)
#pragma unroll
      for (int nf = 0; nf < 2; ++nf)
        acc[mf][nf] = wmma_bf16(a[mf], bb[nf], acc[mf][nf]);
    __syncthreads();
  }

#pragma unroll
  for (int mf = 0; mf < 4; ++mf) {
    const int ep0 = e0 + wm + mf * 16 + half * 8;
#pragma unroll
    for (int nf = 0; nf < 2; ++nf) {
      const int t = t0 + wn + nf * 16 + lr;
      float* dst = out + ((size_t)b * Tn + t) * E + ep0;
      const v8f a8 = acc[mf][nf];
      float4 lo = {a8[0], a8[1], a8[2], a8[3]};
      float4 hi = {a8[4], a8[5], a8[6], a8[7]};
      *(float4*)dst = lo;
      *(float4*)(dst + 4) = hi;
    }
  }
}

// ---------------------------------------------------------------------------
extern "C" void kernel_launch(void* const* d_in, const int* in_sizes, int n_in,
                              void* d_out, int out_size, void* d_ws, size_t ws_size,
                              hipStream_t stream) {
  const float* x     = (const float*)d_in[0];
  const float* wqkv  = (const float*)d_in[1];
  const float* wproj = (const float*)d_in[2];
  float* out = (float*)d_out;
  char* ws = (char*)d_ws;

  bf16_t* xbf   = (bf16_t*)(ws + OFF_XBF);
  bf16_t* wqb   = (bf16_t*)(ws + OFF_WQKV);
  bf16_t* wpb   = (bf16_t*)(ws + OFF_WPROJ);
  bf16_t* Qh    = (bf16_t*)(ws + OFF_QH);
  bf16_t* Kt    = (bf16_t*)(ws + OFF_KT);
  bf16_t* Vt    = (bf16_t*)(ws + OFF_VT);
  bf16_t* Mt    = (bf16_t*)(ws + OFF_MT);
  bf16_t* attc  = (bf16_t*)(ws + OFF_ATT);

  // K0: fp32 -> bf16
  {
    int n4;
    n4 = TOK * E / 4;
    cvt_bf16_kernel<<<(n4 + 255) / 256, 256, 0, stream>>>(x, xbf, n4);
    n4 = E3 * E / 4;
    cvt_bf16_kernel<<<(n4 + 255) / 256, 256, 0, stream>>>(wqkv, wqb, n4);
    n4 = E * E / 4;
    cvt_bf16_kernel<<<(n4 + 255) / 256, 256, 0, stream>>>(wproj, wpb, n4);
  }

  // K1: QKV projection + head-split scatter (TDM-staged)
  qkv_gemm_kernel<<<dim3(E3 / 128, TOK / 128), 256, 0, stream>>>(
      xbf, wqb, Qh, Kt, Vt);

  // K2: per-(b,head) K^T V  (64x64)
  kv_gemm_kernel<<<dim3(Bsz * NH), 256, 0, stream>>>(Kt, Vt, Mt);

  // K3: att = Q M (computed transposed), written in (b,t,n*64+j) order
  att_kernel<<<dim3(Tn / 128, Bsz * NH), 256, 0, stream>>>(Mt, Qh, attc);

  // K4: output projection (TDM-staged), fp32 result straight to d_out
  proj_gemm_kernel<<<dim3(Tn / 128, E / 128, Bsz), 256, 0, stream>>>(
      wpb, attc, out);
}
